// TopKRouter_11579231830571
// MI455X (gfx1250) — matrix operand order repaired and checked by reading, not compiled, not run on hardware
//
#include <hip/hip_runtime.h>
#include <hip/hip_bf16.h>
#include <stdint.h>

typedef __attribute__((ext_vector_type(16))) __bf16 v16bf;
typedef __attribute__((ext_vector_type(8)))  __bf16 v8bf;
typedef __attribute__((ext_vector_type(8)))  float  v8f;
typedef __attribute__((ext_vector_type(4)))  float  v4f;
typedef __attribute__((ext_vector_type(4)))  int    v4i;

#define KDIM   4096
#define NEXP   64
#define KCHUNK 128                  // K per async-staged LDS chunk
#define NCHUNK (KDIM / KCHUNK)      // 32
#define ROWB   (KCHUNK * 2 + 16)    // 272 B per expert row (16B pad: no DS bank conflicts)
#define PLANEB (NEXP * ROWB)        // 17408 B per plane (hi or lo)
#define BUFB   (2 * PLANEB)         // 34816 B per buffer (hi+lo)
#define SMEMB  (2 * BUFB)           // 69632 B double-buffered

typedef __attribute__((address_space(1))) v4i gv4i;   // global int4
typedef __attribute__((address_space(3))) v4i lv4i;   // LDS int4

// ---------------------------------------------------------------------------
// One 16B global->LDS async copy (ASYNCcnt-tracked).
// ---------------------------------------------------------------------------
__device__ __forceinline__ void async_ld16(const void* g, void* l) {
#if __has_builtin(__builtin_amdgcn_global_load_async_to_lds_b128)
  __builtin_amdgcn_global_load_async_to_lds_b128((gv4i*)g, (lv4i*)l, 0, 0);
#else
  asm volatile("global_load_async_to_lds_b128 %0, %1, off"
               :: "v"((uint32_t)(uintptr_t)l), "v"((uint64_t)(uintptr_t)g)
               : "memory");
#endif
}

// ---------------------------------------------------------------------------
// Cooperatively issue one B chunk (64 experts x 128 K, hi+lo planes) into LDS.
// 128 threads x 16 ops x 16B = 32 KB. Padded row stride ROWB.
// ---------------------------------------------------------------------------
__device__ __forceinline__ void issue_b_chunk(const char* gwhiB, const char* gwloB,
                                              char* smem, int buf, int k0, int tid) {
  const int c     = tid & 15;     // 16B column within the 256B K-row
  const int ebase = tid >> 4;     // 0..7
  char* lbase = smem + (size_t)buf * BUFB;
#pragma unroll
  for (int i = 0; i < 16; ++i) {
    const int p = i >> 3;                      // 0 = hi plane, 1 = lo plane
    const int e = (i & 7) * 8 + ebase;         // expert 0..63
    const char* gb = (p ? gwloB : gwhiB) +
                     (size_t)e * (KDIM * 2) + (size_t)k0 * 2 + (size_t)c * 16;
    char* lb = lbase + p * PLANEB + e * ROWB + c * 16;
    async_ld16(gb, lb);
  }
}

// ---------------------------------------------------------------------------
// Pre-kernel: split fp32 gate weights into bf16 (hi, lo) planes in d_ws.
// ---------------------------------------------------------------------------
__global__ void split_gate_kernel(const float* __restrict__ gw,
                                  __bf16* __restrict__ hi,
                                  __bf16* __restrict__ lo, int n) {
  int i = blockIdx.x * blockDim.x + threadIdx.x;
  if (i < n) {
    float w  = gw[i];
    __bf16 h = (__bf16)w;
    hi[i] = h;
    lo[i] = (__bf16)(w - (float)h);
  }
}

// ---------------------------------------------------------------------------
// Router GEMM + top-2 + softmax. 128 threads = 4 waves; wave w owns 16 tokens
// x 64 experts (4 WMMA N-tiles). B staged via double-buffered async LDS copy.
// ---------------------------------------------------------------------------
__global__ __launch_bounds__(128) void router_gemm_kernel(
    const float*  __restrict__ x,
    const __bf16* __restrict__ gwhi,
    const __bf16* __restrict__ gwlo,
    float*        __restrict__ logits,
    int*          __restrict__ topk_idx,
    float*        __restrict__ topk_p) {
  __shared__ __attribute__((aligned(16))) char smem[SMEMB];

  const int tid    = threadIdx.x;
  const int wave   = tid >> 5;
  const int lane   = tid & 31;
  const int laneM  = lane & 15;
  const int laneHi = lane >> 4;
  const int tokenBase = blockIdx.x * 64;
  const int row = tokenBase + wave * 16 + laneM;

  const float* xrow   = x + (size_t)row * KDIM;
  const char*  gwhiB  = (const char*)gwhi;
  const char*  gwloB  = (const char*)gwlo;

  v8f acc[4] = {};

  issue_b_chunk(gwhiB, gwloB, smem, 0, 0, tid);    // preload chunk 0

  for (int chunk = 0; chunk < NCHUNK; ++chunk) {
    const int k0c = chunk * KCHUNK;
    if (chunk + 1 < NCHUNK) {
      issue_b_chunk(gwhiB, gwloB, smem, (chunk + 1) & 1, k0c + KCHUNK, tid);
      __builtin_prefetch(xrow + k0c + KCHUNK, 0, 1);     // warm next x chunk
      asm volatile("s_wait_asynccnt 0x10" ::: "memory"); // chunk `chunk` done
    } else {
      asm volatile("s_wait_asynccnt 0x0" ::: "memory");
    }
    __syncthreads();                                     // buffer valid WG-wide

    const char* bbuf = smem + (chunk & 1) * BUFB;

#pragma unroll
    for (int ks = 0; ks < KCHUNK / 32; ++ks) {
      const int k = k0c + ks * 32;

      // ---- A fragment (ISA 16-bit A 16x32 layout), fp32 -> bf16 hi/lo ----
      const int kb = k + laneHi * 8;
      v4f f0 = *(const v4f*)(xrow + kb);
      v4f f1 = *(const v4f*)(xrow + kb + 4);
      v4f f2 = *(const v4f*)(xrow + kb + 16);
      v4f f3 = *(const v4f*)(xrow + kb + 20);
      float av[16] = {f0.x, f0.y, f0.z, f0.w, f1.x, f1.y, f1.z, f1.w,
                      f2.x, f2.y, f2.z, f2.w, f3.x, f3.y, f3.z, f3.w};
      v16bf a_hi, a_lo;
#pragma unroll
      for (int i = 0; i < 16; ++i) {
        float v  = av[i];
        __bf16 h = (__bf16)v;
        a_hi[i] = h;
        a_lo[i] = (__bf16)(v - (float)h);
      }

      // ---- B fragments from LDS + 3-term split WMMA per N-tile ----
      const int koff = ks * 64 + laneHi * 16;     // byte offset within row
#pragma unroll
      for (int t = 0; t < 4; ++t) {
        const int e   = t * 16 + laneM;
        const int off = e * ROWB + koff;
        v8bf bh0 = *(const v8bf*)(bbuf + off);
        v8bf bh1 = *(const v8bf*)(bbuf + off + 32);
        v8bf bl0 = *(const v8bf*)(bbuf + PLANEB + off);
        v8bf bl1 = *(const v8bf*)(bbuf + PLANEB + off + 32);
        v16bf b_hi, b_lo;
#pragma unroll
        for (int i = 0; i < 8; ++i) {
          b_hi[i] = bh0[i]; b_hi[8 + i] = bh1[i];
          b_lo[i] = bl0[i]; b_lo[8 + i] = bl1[i];
        }
        acc[t] = __builtin_amdgcn_wmma_f32_16x16x32_bf16(
            false, a_hi, false, b_hi, (short)0, acc[t], false, false);
        acc[t] = __builtin_amdgcn_wmma_f32_16x16x32_bf16(
            false, a_hi, false, b_lo, (short)0, acc[t], false, false);
        acc[t] = __builtin_amdgcn_wmma_f32_16x16x32_bf16(
            false, a_lo, false, b_hi, (short)0, acc[t], false, false);
      }
    }
    __syncthreads();                 // all waves done reading this buffer
  }

  // ---- epilogue: alias the B staging LDS as a 64x64 logits tile ----
  float* flds = (float*)smem;        // stride 65 -> conflict-free scans
#pragma unroll
  for (int t = 0; t < 4; ++t) {
#pragma unroll
    for (int r = 0; r < 8; ++r) {
      int tok = wave * 16 + r + laneHi * 8;
      int e   = t * 16 + laneM;
      flds[tok * 65 + e] = acc[t][r];
    }
  }
  __syncthreads();

  // coalesced logits store
  for (int j = tid; j < 64 * 64; j += 128) {
    int tok = j >> 6;
    int e   = j & 63;
    logits[(size_t)(tokenBase + tok) * NEXP + e] = flds[tok * 65 + e];
  }

  // top-2 + softmax, one thread per token
  if (tid < 64) {
    const float* rowp = &flds[tid * 65];
    float v0 = -3.402823466e38f, v1 = -3.402823466e38f;
    int   i0 = 0, i1 = 0;
#pragma unroll 8
    for (int e = 0; e < 64; ++e) {
      float v = rowp[e];
      if (v > v0)      { v1 = v0; i1 = i0; v0 = v; i0 = e; }
      else if (v > v1) { v1 = v;  i1 = e; }
    }
    float e1  = __expf(v1 - v0);
    float inv = 1.0f / (1.0f + e1);
    int tok = tokenBase + tid;
    topk_idx[tok * 2 + 0] = i0;
    topk_idx[tok * 2 + 1] = i1;
    topk_p[tok * 2 + 0]   = inv;
    topk_p[tok * 2 + 1]   = e1 * inv;
  }
}

extern "C" void kernel_launch(void* const* d_in, const int* in_sizes, int n_in,
                              void* d_out, int out_size, void* d_ws, size_t ws_size,
                              hipStream_t stream) {
  const float* x  = (const float*)d_in[0];   // [4,4096,4096] f32
  const float* gw = (const float*)d_in[1];   // [64,4096] f32

  const int gate_n  = in_sizes[1];           // 64*4096
  const int mtokens = in_sizes[0] / KDIM;    // 16384

  __bf16* gwhi = (__bf16*)d_ws;
  __bf16* gwlo = gwhi + gate_n;

  split_gate_kernel<<<(gate_n + 255) / 256, 256, 0, stream>>>(gw, gwhi, gwlo,
                                                              gate_n);

  float* logits   = (float*)d_out;
  int*   topk_idx = (int*)(logits + (size_t)mtokens * NEXP);
  float* topk_p   = (float*)(topk_idx + (size_t)mtokens * 2);

  router_gemm_kernel<<<dim3(mtokens / 64), dim3(128), 0, stream>>>(
      x, gwhi, gwlo, logits, topk_idx, topk_p);
}
